// PerceiverCrossAttention_58317065945679
// MI455X (gfx1250) — matrix operand order repaired and checked by reading, not compiled, not run on hardware
//
#include <hip/hip_runtime.h>
#include <hip/hip_bf16.h>

// ---------------- model constants ----------------
#define BB   128
#define NN   256
#define DM   1024
#define HH   8
#define DKh  128
#define KK   4
#define NLAYER 2
#define ATT_SCALE 0.08838834764831845f   // 1/sqrt(128)

// epilogue flags (template parameter)
#define F_GELU 1
#define F_RES  2
#define F_OBF  4   // write bf16 output
#define F_VT   8   // head-transposed bf16 store (for V)

// ---------------- vector types ----------------
typedef __attribute__((ext_vector_type(16))) __bf16 v16bf;
typedef __attribute__((ext_vector_type(8)))  __bf16 v8bf;
typedef __attribute__((ext_vector_type(8)))  float  v8f;

__device__ __forceinline__ unsigned short f2bfraw(float f) {
  union { float f; unsigned u; } c; c.f = f;
  unsigned r = c.u + 0x7FFFu + ((c.u >> 16) & 1u);   // round-to-nearest-even
  return (unsigned short)(r >> 16);
}
__device__ __forceinline__ __bf16 f2bf(float f) {
  unsigned short h = f2bfraw(f);
  return __builtin_bit_cast(__bf16, h);
}

#define LDSROW 80   // bf16 row stride in LDS (160B; 16B-aligned segments)

// issue one 128x64 bf16 tile pair (A,B) as 8 async b128 chunks per thread
__device__ __forceinline__ void stage_tiles(
    const __bf16* __restrict__ Ab, long lda,
    const __bf16* __restrict__ Bb, long ldb,
    int kc, __bf16* al, __bf16* bl, int tid)
{
  #pragma unroll
  for (int it = 0; it < 4; ++it) {
    int lin = tid + it * 256;            // 0..1023 16B chunks per tile
    int row = lin >> 3;                  // 0..127
    int seg = (lin & 7) << 3;            // 0..56 (bf16 units)
    const __bf16* ga = Ab + (long)row * lda + kc + seg;
    unsigned la = (unsigned)(size_t)(al + row * LDSROW + seg);
    asm volatile("global_load_async_to_lds_b128 %0, %1, off"
                 :: "v"(la), "v"(ga) : "memory");
    const __bf16* gb = Bb + (long)row * ldb + kc + seg;
    unsigned lb = (unsigned)(size_t)(bl + row * LDSROW + seg);
    asm volatile("global_load_async_to_lds_b128 %0, %1, off"
                 :: "v"(lb), "v"(gb) : "memory");
  }
}

// =======================================================================
// bf16-WMMA GEMM, double-buffered async-to-LDS pipeline.
//   A : M x Kd bf16 row-major (stride lda), batched via sAb/sAh
//   B : Nout x Kd bf16 row-major [n][k] (stride ldb), batched via sBb/sBh
//   C : f32 (default) / bf16 (F_OBF) / head-transposed bf16 (F_VT)
// Block tile 128x128, 8 waves (2x4), wave tile 64x32 = 4x2 WMMA 16x16x32.
// Stage i+1 DMA overlaps stage i compute; async loads retire in order, so
// s_wait_asynccnt 8 (= next stage still outstanding) gates the current one.
// =======================================================================
template <int FLAGS>
__global__ __launch_bounds__(256) void k_gemm(
    const __bf16* __restrict__ A,  long lda, long sAb, long sAh,
    const __bf16* __restrict__ Bm, long ldb, long sBb, long sBh,
    const float* __restrict__ bias,
    const float* __restrict__ Rr, long ldr,
    void* __restrict__ Cp,  long ldc, long sCb, long sCh,
    int Kd, float alpha, int Hz)
{
  __shared__ __bf16 Al[2][128 * LDSROW];
  __shared__ __bf16 Bl[2][128 * LDSROW];

  const int tid  = threadIdx.x;
  const int lane = tid & 31, wid = tid >> 5;
  const int wm = wid >> 2, wn = wid & 3;       // wave grid: 2 (M) x 4 (N)
  const int hi = lane >> 4, l15 = lane & 15;
  const int bx = blockIdx.x, by = blockIdx.y;
  const int zb = blockIdx.z / Hz, zh = blockIdx.z % Hz;

  const __bf16* Ab = A  + (long)zb * sAb + (long)zh * sAh + (long)(by * 128) * lda;
  const __bf16* Bb = Bm + (long)zb * sBb + (long)zh * sBh + (long)(bx * 128) * ldb;

  v8f acc[4][2];
  #pragma unroll
  for (int i = 0; i < 4; ++i)
    #pragma unroll
    for (int j = 0; j < 2; ++j)
      #pragma unroll
      for (int r = 0; r < 8; ++r) acc[i][j][r] = 0.0f;

  const int nst = Kd >> 6;                     // 64-wide K stages
  stage_tiles(Ab, lda, Bb, ldb, 0, Al[0], Bl[0], tid);

  for (int st = 0; st < nst; ++st) {
    const int cur = st & 1;
    if (st + 1 < nst) {
      stage_tiles(Ab, lda, Bb, ldb, (st + 1) << 6, Al[cur ^ 1], Bl[cur ^ 1], tid);
      asm volatile("s_wait_asynccnt 8" ::: "memory");   // current stage resident
    } else {
      asm volatile("s_wait_asynccnt 0" ::: "memory");
    }
    __syncthreads();

    const __bf16* Ac = Al[cur];
    const __bf16* Bc = Bl[cur];
    #pragma unroll
    for (int ks = 0; ks < 64; ks += 32) {
      v16bf af[4];
      #pragma unroll
      for (int i = 0; i < 4; ++i) {
        // A 16x32 bf16 operand: lane(l15,hi): row=l15, K = hi*8 + {0..7,16..23}
        const __bf16* ap = Ac + (wm * 64 + i * 16 + l15) * LDSROW + ks + hi * 8;
        v8bf p0 = *(const v8bf*)ap;
        v8bf p1 = *(const v8bf*)(ap + 16);
        af[i] = __builtin_shufflevector(p0, p1, 0,1,2,3,4,5,6,7,8,9,10,11,12,13,14,15);
      }
      v16bf bfr[2];
      #pragma unroll
      for (int j = 0; j < 2; ++j) {
        // B 32x16 bf16 operand: lane(l15,hi): col=l15, K = hi*8 + {0..7,16..23}
        const __bf16* bp = Bc + (wn * 32 + j * 16 + l15) * LDSROW + ks + hi * 8;
        v8bf p0 = *(const v8bf*)bp;
        v8bf p1 = *(const v8bf*)(bp + 16);
        bfr[j] = __builtin_shufflevector(p0, p1, 0,1,2,3,4,5,6,7,8,9,10,11,12,13,14,15);
      }
      #pragma unroll
      for (int i = 0; i < 4; ++i)
        #pragma unroll
        for (int j = 0; j < 2; ++j)
          acc[i][j] = __builtin_amdgcn_wmma_f32_16x16x32_bf16(
              false, af[i], false, bfr[j], (short)0, acc[i][j], false, false);
    }
    __syncthreads();
  }

  // ---- epilogue: D layout: VGPR r -> M = r + hi*8, N = l15 ----
  const long zC = (long)zb * sCb + (long)zh * sCh;
  #pragma unroll
  for (int i = 0; i < 4; ++i) {
    #pragma unroll
    for (int j = 0; j < 2; ++j) {
      int n = bx * 128 + wn * 32 + j * 16 + l15;
      float bv = bias ? bias[n] : 0.0f;
      #pragma unroll
      for (int r = 0; r < 8; ++r) {
        int m = by * 128 + wm * 64 + i * 16 + hi * 8 + r;
        float v = acc[i][j][r] * alpha + bv;
        if (FLAGS & F_GELU) v = 0.5f * v * (1.0f + erff(v * 0.70710678118654752f));
        if (FLAGS & F_RES)  v += Rr[(long)m * ldr + n];
        if (FLAGS & F_VT) {
          // m = b*256+token, n = h*128+dk -> Vt[b][h][dk][token]
          long idx = (((long)(m >> 8) * HH + (n >> 7)) * DKh + (n & 127)) * NN + (m & 255);
          ((__bf16*)Cp)[idx] = f2bf(v);
        } else if (FLAGS & F_OBF) {
          ((__bf16*)Cp)[zC + (long)m * ldc + n] = f2bf(v);
        } else {
          ((float*)Cp)[zC + (long)m * ldc + n] = v;
        }
      }
    }
  }
}

// =======================================================================
// Weight convert + transpose: f32 [Kd][Nout] -> bf16 [Nout][Kd]
// =======================================================================
__global__ __launch_bounds__(256) void k_w2bf(
    const float* __restrict__ w, __bf16* __restrict__ o, int Kd, int Nout)
{
  __shared__ float t[32][33];
  int n0 = blockIdx.x * 32, k0 = blockIdx.y * 32;
  int c = threadIdx.x & 31, r = threadIdx.x >> 5;   // r: 0..7
  #pragma unroll
  for (int rr = 0; rr < 4; ++rr)
    t[r + rr * 8][c] = w[(long)(k0 + r + rr * 8) * Nout + n0 + c];
  __syncthreads();
  #pragma unroll
  for (int rr = 0; rr < 4; ++rr)
    o[(long)(n0 + r + rr * 8) * Kd + k0 + c] = f2bf(t[c][r + rr * 8]);
}

// =======================================================================
// LayerNorm over D=1024 -> bf16 output. One block (256 thr) per row.
// =======================================================================
__global__ __launch_bounds__(256) void k_layernorm(
    const float* __restrict__ x, const float* __restrict__ g,
    const float* __restrict__ b, __bf16* __restrict__ y)
{
  __shared__ float s1[8], s2[8];
  long row = blockIdx.x;
  int c = threadIdx.x * 4;
  float4 v = *(const float4*)(x + row * DM + c);
  float s = v.x + v.y + v.z + v.w;
  float q = v.x*v.x + v.y*v.y + v.z*v.z + v.w*v.w;
  #pragma unroll
  for (int o = 16; o; o >>= 1) { s += __shfl_xor(s, o); q += __shfl_xor(q, o); }
  if ((threadIdx.x & 31) == 0) { s1[threadIdx.x >> 5] = s; s2[threadIdx.x >> 5] = q; }
  __syncthreads();
  float ts = 0.f, tq = 0.f;
  #pragma unroll
  for (int i = 0; i < 8; ++i) { ts += s1[i]; tq += s2[i]; }
  float mean = ts * (1.0f / DM);
  float var  = tq * (1.0f / DM) - mean * mean;
  float inv  = rsqrtf(var + 1e-5f);
  unsigned short h0 = f2bfraw((v.x - mean) * inv * g[c+0] + b[c+0]);
  unsigned short h1 = f2bfraw((v.y - mean) * inv * g[c+1] + b[c+1]);
  unsigned short h2 = f2bfraw((v.z - mean) * inv * g[c+2] + b[c+2]);
  unsigned short h3 = f2bfraw((v.w - mean) * inv * g[c+3] + b[c+3]);
  uint2 u; u.x = (unsigned)h0 | ((unsigned)h1 << 16);
  u.y = (unsigned)h2 | ((unsigned)h3 << 16);
  *(uint2*)(y + row * DM + c) = u;
}

// =======================================================================
// Row softmax over 256 f32 scores -> bf16 probs. One block per row.
// =======================================================================
__global__ __launch_bounds__(256) void k_softmax256(
    const float* __restrict__ s, __bf16* __restrict__ pout)
{
  __shared__ float sh[8];
  const float* p = s + (long)blockIdx.x * 256;
  int tid = threadIdx.x;
  float v = p[tid];
  float m = v;
  #pragma unroll
  for (int o = 16; o; o >>= 1) m = fmaxf(m, __shfl_xor(m, o));
  if ((tid & 31) == 0) sh[tid >> 5] = m;
  __syncthreads();
  float mm = sh[0];
  #pragma unroll
  for (int i = 1; i < 8; ++i) mm = fmaxf(mm, sh[i]);
  __syncthreads();
  float e = expf(v - mm);
  float su = e;
  #pragma unroll
  for (int o = 16; o; o >>= 1) su += __shfl_xor(su, o);
  if ((tid & 31) == 0) sh[tid >> 5] = su;
  __syncthreads();
  float ts = 0.f;
  #pragma unroll
  for (int i = 0; i < 8; ++i) ts += sh[i];
  pout[(long)blockIdx.x * 256 + tid] = f2bf(e / ts);
}

// =======================================================================
// Encoder cross-attention (4 keys + temporal bias). One wave per (b,q,h).
// =======================================================================
__global__ __launch_bounds__(256) void k_cross_attn(
    const float* __restrict__ q,  const float* __restrict__ kc,
    const float* __restrict__ vc, const float* __restrict__ tb,
    __bf16* __restrict__ outc, float* __restrict__ encA)
{
  int gw   = blockIdx.x * 8 + (threadIdx.x >> 5);
  int lane = threadIdx.x & 31;
  int h  = gw % HH;
  int qn = (gw / HH) % NN;
  int b  = gw / (HH * NN);

  float4 qv = *(const float4*)(q + ((long)b * NN + qn) * DM + h * DKh + lane * 4);
  float sc[KK];
  #pragma unroll
  for (int j = 0; j < KK; ++j) {
    float4 kv = *(const float4*)(kc + ((long)b * KK + j) * DM + h * DKh + lane * 4);
    float d = qv.x*kv.x + qv.y*kv.y + qv.z*kv.z + qv.w*kv.w;
    #pragma unroll
    for (int o = 16; o; o >>= 1) d += __shfl_xor(d, o);
    sc[j] = d * ATT_SCALE + tb[b * KK + j];
  }
  float m = fmaxf(fmaxf(sc[0], sc[1]), fmaxf(sc[2], sc[3]));
  float e0 = expf(sc[0] - m), e1 = expf(sc[1] - m);
  float e2 = expf(sc[2] - m), e3 = expf(sc[3] - m);
  float inv = 1.0f / (e0 + e1 + e2 + e3);
  float p0 = e0 * inv, p1 = e1 * inv, p2 = e2 * inv, p3 = e3 * inv;

  const float* vb = vc + (long)b * KK * DM + h * DKh + lane * 4;
  float4 v0 = *(const float4*)(vb + 0 * DM);
  float4 v1 = *(const float4*)(vb + 1 * DM);
  float4 v2 = *(const float4*)(vb + 2 * DM);
  float4 v3 = *(const float4*)(vb + 3 * DM);
  unsigned short h0 = f2bfraw(p0*v0.x + p1*v1.x + p2*v2.x + p3*v3.x);
  unsigned short h1 = f2bfraw(p0*v0.y + p1*v1.y + p2*v2.y + p3*v3.y);
  unsigned short h2 = f2bfraw(p0*v0.z + p1*v1.z + p2*v2.z + p3*v3.z);
  unsigned short h3 = f2bfraw(p0*v0.w + p1*v1.w + p2*v2.w + p3*v3.w);
  uint2 u; u.x = (unsigned)h0 | ((unsigned)h1 << 16);
  u.y = (unsigned)h2 | ((unsigned)h3 << 16);
  *(uint2*)(outc + ((long)b * NN + qn) * DM + h * DKh + lane * 4) = u;

  if (lane < KK) {
    float pv = (lane == 0) ? p0 : (lane == 1) ? p1 : (lane == 2) ? p2 : p3;
    encA[(((long)b * HH + h) * NN + qn) * KK + lane] = pv;
  }
}

// =======================================================================
// Decoder query projection: y = x(bf16) @ W(f32) + bias
// =======================================================================
__global__ void k_matvec(const __bf16* __restrict__ x, const float* __restrict__ W,
                         const float* __restrict__ bias, float* __restrict__ y)
{
  int col = blockIdx.x * 256 + threadIdx.x;
  float s = bias[col];
  for (int k = 0; k < DM; ++k) s += (float)x[k] * W[(long)k * DM + col];
  y[col] = s;
}

// =======================================================================
// Decoder attention: 1 query, 256 keys. One block (256 thr) per (b,h).
// =======================================================================
__global__ __launch_bounds__(256) void k_dec_attn(
    const float* __restrict__ qv, const float* __restrict__ kd,
    const float* __restrict__ vd, __bf16* __restrict__ dctx)
{
  __shared__ float sq[DKh];
  __shared__ float sp[NN];
  __shared__ float sh[8];
  int b = blockIdx.x / HH, h = blockIdx.x % HH;
  int tid = threadIdx.x;
  if (tid < DKh) sq[tid] = qv[h * DKh + tid];
  __syncthreads();

  const float* kp = kd + ((long)b * NN + tid) * DM + h * DKh;
  float s = 0.f;
  for (int d = 0; d < DKh; ++d) s += sq[d] * kp[d];
  s *= ATT_SCALE;

  float m = s;
  #pragma unroll
  for (int o = 16; o; o >>= 1) m = fmaxf(m, __shfl_xor(m, o));
  if ((tid & 31) == 0) sh[tid >> 5] = m;
  __syncthreads();
  float mm = sh[0];
  #pragma unroll
  for (int i = 1; i < 8; ++i) mm = fmaxf(mm, sh[i]);
  __syncthreads();
  float e = expf(s - mm);
  float su = e;
  #pragma unroll
  for (int o = 16; o; o >>= 1) su += __shfl_xor(su, o);
  if ((tid & 31) == 0) sh[tid >> 5] = su;
  __syncthreads();
  float ts = 0.f;
  #pragma unroll
  for (int i = 0; i < 8; ++i) ts += sh[i];
  sp[tid] = e / ts;
  __syncthreads();

  if (tid < DKh) {
    float o = 0.f;
    for (int j = 0; j < NN; ++j)
      o += sp[j] * vd[((long)b * NN + j) * DM + h * DKh + tid];
    dctx[(long)b * DM + h * DKh + tid] = f2bf(o);
  }
}

// =======================================================================
// Host orchestration
// =======================================================================
extern "C" void kernel_launch(void* const* d_in, const int* in_sizes, int n_in,
                              void* d_out, int out_size, void* d_ws, size_t ws_size,
                              hipStream_t stream) {
  (void)in_sizes; (void)n_in; (void)out_size; (void)ws_size;

  const float* LAT_IN = (const float*)d_in[0];
  const float* MOD_IN = (const float*)d_in[1];
  const float* TBIAS  = (const float*)d_in[2];
  auto P = [&](int i) { return (const float*)d_in[i]; };
  // JAX pytree flatten (dict keys sorted, recursive):
  const int I_DEC_K_B=3, I_DEC_K_W=4, I_DEC_O_B=5, I_DEC_O_W=6;
  const int I_DEC_Q_B=7, I_DEC_Q_W=8, I_DEC_V_B=9, I_DEC_V_W=10;
  const int I_DEC_NLAT_B=11, I_DEC_NLAT_G=12, I_DEC_NQ_B=13, I_DEC_NQ_G=14, I_DEC_QUERY=15;
  const int I_ENC_K_B=16, I_ENC_K_W=17, I_ENC_O_B=18, I_ENC_O_W=19;
  const int I_ENC_Q_B=20, I_ENC_Q_W=21, I_ENC_V_B=22, I_ENC_V_W=23;
  const int I_EFF_L1_B=24, I_EFF_L1_W=25, I_EFF_L2_B=26, I_EFF_L2_W=27;
  const int I_EFF_LN_B=28, I_EFF_LN_G=29;
  const int I_ENC_NIN_B=30, I_ENC_NIN_G=31, I_ENC_NLAT_B=32, I_ENC_NLAT_G=33;
  const int I_LAYER0=34;   // per layer: k.b,k.w,o.b,o.w,q.b,q.w,v.b,v.w,l1.b,l1.w,l2.b,l2.w,ln.b,ln.g,norm.b,norm.g

  // ---- workspace layout ----
  const size_t SZ = (size_t)BB * NN * DM;
  float*  scor = (float*)d_ws;                      // 2*SZ   f32 scores
  float*  qb   = scor + 2 * SZ;                     // SZ     f32 Q (cross)
  float*  kbuf = qb + SZ;                           // SZ
  float*  vbuf = kbuf + SZ;                         // SZ
  float*  qvec = vbuf + SZ;                         // DM
  __bf16* tmpbf = (__bf16*)(qvec + DM);             // SZ     LN output
  __bf16* ninbf = tmpbf + SZ;                       // B*K*D
  __bf16* qbf   = ninbf + (size_t)BB * KK * DM;     // SZ
  __bf16* kbf   = qbf + SZ;                         // SZ
  __bf16* vtbf  = kbf + SZ;                         // SZ (head-transposed V)
  __bf16* pbf   = vtbf + SZ;                        // 2*SZ   softmax probs
  __bf16* aobf  = pbf + 2 * SZ;                     // SZ     attention context
  __bf16* hidbf = aobf + SZ;                        // 4*SZ   FFN hidden
  __bf16* nq1bf = hidbf + 4 * SZ;                   // DM
  __bf16* dctxbf = nq1bf + DM;                      // B*DM
  __bf16* warena = dctxbf + (size_t)BB * DM;        // bf16 weights
  size_t  woff = 0;

  float* lat   = (float*)d_out;                     // latents output buffer
  float* fused = lat + SZ;
  float* encA  = fused + (size_t)BB * DM;

  auto conv = [&](const float* w, int Kd, int Nout) {
    __bf16* dst = warena + woff; woff += (size_t)Kd * Nout;
    k_w2bf<<<dim3(Nout / 32, Kd / 32), dim3(256), 0, stream>>>(w, dst, Kd, Nout);
    return dst;
  };
  auto gemm = [&](const __bf16* A, long lda, long sAb, long sAh,
                  const __bf16* Bm, long ldb, long sBb, long sBh,
                  const float* bias, const float* Rr, long ldr,
                  void* Cp, long ldc, long sCb, long sCh,
                  int M, int Nout, int Kd, float alpha, int flags, int Hz, int nz) {
    dim3 g(Nout / 128, M / 128, nz);
    dim3 blk(256);
    switch (flags) {
      case 0:
        k_gemm<0><<<g, blk, 0, stream>>>(A, lda, sAb, sAh, Bm, ldb, sBb, sBh,
                                         bias, Rr, ldr, Cp, ldc, sCb, sCh, Kd, alpha, Hz);
        break;
      case F_RES:
        k_gemm<F_RES><<<g, blk, 0, stream>>>(A, lda, sAb, sAh, Bm, ldb, sBb, sBh,
                                             bias, Rr, ldr, Cp, ldc, sCb, sCh, Kd, alpha, Hz);
        break;
      case F_OBF:
        k_gemm<F_OBF><<<g, blk, 0, stream>>>(A, lda, sAb, sAh, Bm, ldb, sBb, sBh,
                                             bias, Rr, ldr, Cp, ldc, sCb, sCh, Kd, alpha, Hz);
        break;
      case F_OBF | F_VT:
        k_gemm<F_OBF | F_VT><<<g, blk, 0, stream>>>(A, lda, sAb, sAh, Bm, ldb, sBb, sBh,
                                                    bias, Rr, ldr, Cp, ldc, sCb, sCh, Kd, alpha, Hz);
        break;
      default: // F_GELU | F_OBF
        k_gemm<F_GELU | F_OBF><<<g, blk, 0, stream>>>(A, lda, sAb, sAh, Bm, ldb, sBb, sBh,
                                                      bias, Rr, ldr, Cp, ldc, sCb, sCh, Kd, alpha, Hz);
        break;
    }
  };

  // ---- weight conversion (f32 [Kd][Nout] -> bf16 [Nout][Kd]) ----
  __bf16* WQe = conv(P(I_ENC_Q_W), DM, DM);
  __bf16* WKe = conv(P(I_ENC_K_W), DM, DM);
  __bf16* WVe = conv(P(I_ENC_V_W), DM, DM);
  __bf16* WOe = conv(P(I_ENC_O_W), DM, DM);
  __bf16* W1e = conv(P(I_EFF_L1_W), DM, 4 * DM);
  __bf16* W2e = conv(P(I_EFF_L2_W), 4 * DM, DM);
  __bf16 *WQl[NLAYER], *WKl[NLAYER], *WVl[NLAYER], *WOl[NLAYER], *W1l[NLAYER], *W2l[NLAYER];
  for (int li = 0; li < NLAYER; ++li) {
    int base = I_LAYER0 + li * 16;
    WKl[li] = conv(P(base + 1), DM, DM);
    WOl[li] = conv(P(base + 3), DM, DM);
    WQl[li] = conv(P(base + 5), DM, DM);
    WVl[li] = conv(P(base + 7), DM, DM);
    W1l[li] = conv(P(base + 9), DM, 4 * DM);
    W2l[li] = conv(P(base + 11), 4 * DM, DM);
  }
  __bf16* WKd = conv(P(I_DEC_K_W), DM, DM);
  __bf16* WVd = conv(P(I_DEC_V_W), DM, DM);
  __bf16* WOd = conv(P(I_DEC_O_W), DM, DM);

  // ================= encoder cross-attention =================
  k_layernorm<<<BB * NN, 256, 0, stream>>>(LAT_IN, P(I_ENC_NLAT_G), P(I_ENC_NLAT_B), tmpbf);
  k_layernorm<<<BB * KK, 256, 0, stream>>>(MOD_IN, P(I_ENC_NIN_G),  P(I_ENC_NIN_B),  ninbf);
  gemm(tmpbf, DM,0,0, WQe, DM,0,0, P(I_ENC_Q_B), nullptr,0, qb,   DM,0,0, BB*NN, DM, DM, 1.f, 0, 1, 1);
  gemm(ninbf, DM,0,0, WKe, DM,0,0, P(I_ENC_K_B), nullptr,0, kbuf, DM,0,0, BB*KK, DM, DM, 1.f, 0, 1, 1);
  gemm(ninbf, DM,0,0, WVe, DM,0,0, P(I_ENC_V_B), nullptr,0, vbuf, DM,0,0, BB*KK, DM, DM, 1.f, 0, 1, 1);
  k_cross_attn<<<BB * NN * HH / 8, 256, 0, stream>>>(qb, kbuf, vbuf, TBIAS, aobf, encA);
  gemm(aobf, DM,0,0, WOe, DM,0,0, P(I_ENC_O_B), LAT_IN, DM, lat, DM,0,0, BB*NN, DM, DM, 1.f, F_RES, 1, 1);
  // encoder FFN
  k_layernorm<<<BB * NN, 256, 0, stream>>>(lat, P(I_EFF_LN_G), P(I_EFF_LN_B), tmpbf);
  gemm(tmpbf, DM,0,0, W1e, DM,0,0, P(I_EFF_L1_B), nullptr,0, hidbf, 4*DM,0,0, BB*NN, 4*DM, DM,   1.f, F_GELU|F_OBF, 1, 1);
  gemm(hidbf, 4*DM,0,0, W2e, 4*DM,0,0, P(I_EFF_L2_B), lat, DM, lat, DM,0,0,   BB*NN, DM,   4*DM, 1.f, F_RES, 1, 1);

  // ================= self-attention layers =================
  for (int li = 0; li < NLAYER; ++li) {
    int base = I_LAYER0 + li * 16;
    const float *KB = P(base+0), *OB = P(base+2), *QB = P(base+4), *VB = P(base+6);
    const float *L1B = P(base+8), *L2B = P(base+10);
    const float *FLNB = P(base+12), *FLNG = P(base+13);
    const float *NB = P(base+14), *NG = P(base+15);

    k_layernorm<<<BB * NN, 256, 0, stream>>>(lat, NG, NB, tmpbf);
    gemm(tmpbf, DM,0,0, WQl[li], DM,0,0, QB, nullptr,0, qbf,  DM,0,0, BB*NN, DM, DM, 1.f, F_OBF, 1, 1);
    gemm(tmpbf, DM,0,0, WKl[li], DM,0,0, KB, nullptr,0, kbf,  DM,0,0, BB*NN, DM, DM, 1.f, F_OBF, 1, 1);
    gemm(tmpbf, DM,0,0, WVl[li], DM,0,0, VB, nullptr,0, vtbf, 0,0,0,  BB*NN, DM, DM, 1.f, F_OBF|F_VT, 1, 1);
    // scores[b,h] = SCALE * Q @ K^T   (K already [key][dk] row-major)
    gemm(qbf, DM, (long)NN*DM, DKh,
         kbf, DM, (long)NN*DM, DKh,
         nullptr, nullptr, 0,
         scor, NN, (long)HH*NN*NN, (long)NN*NN,
         NN, NN, DKh, ATT_SCALE, 0, HH, BB*HH);
    k_softmax256<<<BB * HH * NN, 256, 0, stream>>>(scor, pbf);
    // context[b,h] = P @ V   (V stored head-transposed [b][h][dk][token])
    gemm(pbf, NN, (long)HH*NN*NN, (long)NN*NN,
         vtbf, NN, (long)HH*DKh*NN, (long)DKh*NN,
         nullptr, nullptr, 0,
         aobf, DM, (long)NN*DM, DKh,
         NN, DKh, NN, 1.f, F_OBF, HH, BB*HH);
    gemm(aobf, DM,0,0, WOl[li], DM,0,0, OB, lat, DM, lat, DM,0,0, BB*NN, DM, DM, 1.f, F_RES, 1, 1);
    // FFN
    k_layernorm<<<BB * NN, 256, 0, stream>>>(lat, FLNG, FLNB, tmpbf);
    gemm(tmpbf, DM,0,0, W1l[li], DM,0,0, L1B, nullptr,0, hidbf, 4*DM,0,0, BB*NN, 4*DM, DM,   1.f, F_GELU|F_OBF, 1, 1);
    gemm(hidbf, 4*DM,0,0, W2l[li], 4*DM,0,0, L2B, lat, DM, lat, DM,0,0,   BB*NN, DM,   4*DM, 1.f, F_RES, 1, 1);
  }

  // ================= decoder =================
  k_layernorm<<<1, 256, 0, stream>>>(P(I_DEC_QUERY), P(I_DEC_NQ_G), P(I_DEC_NQ_B), nq1bf);
  k_matvec<<<DM / 256, 256, 0, stream>>>(nq1bf, P(I_DEC_Q_W), P(I_DEC_Q_B), qvec);
  k_layernorm<<<BB * NN, 256, 0, stream>>>(lat, P(I_DEC_NLAT_G), P(I_DEC_NLAT_B), tmpbf);
  gemm(tmpbf, DM,0,0, WKd, DM,0,0, P(I_DEC_K_B), nullptr,0, kbuf, DM,0,0, BB*NN, DM, DM, 1.f, 0, 1, 1);
  gemm(tmpbf, DM,0,0, WVd, DM,0,0, P(I_DEC_V_B), nullptr,0, vbuf, DM,0,0, BB*NN, DM, DM, 1.f, 0, 1, 1);
  k_dec_attn<<<BB * HH, 256, 0, stream>>>(qvec, kbuf, vbuf, dctxbf);
  gemm(dctxbf, DM,0,0, WOd, DM,0,0, P(I_DEC_O_B), nullptr,0, fused, DM,0,0, BB, DM, DM, 1.f, 0, 1, 1);
}
